// ObjectBiasedStateTokenizerV42_11141145165865
// MI455X (gfx1250) — matrix operand order repaired and checked by reading, compiled, not run on hardware
//
#include <hip/hip_runtime.h>
#include <hip/hip_bf16.h>
#include <math.h>

// ---------------------------------------------------------------------------
// ObjectBiasedStateTokenizer for MI455X (gfx1250, wave32, WMMA + async-LDS)
// All big GEMMs: bf16 operands, fp32 accumulate via v_wmma_f32_16x16x32_bf16.
// Operand tiles staged into LDS with global_load_async_to_lds_b128 (ASYNCcnt),
// double-buffered so the DMA overlaps the WMMA stream. Inputs are pre-packed
// (concat + fp32->bf16) once; weights pre-transposed to bf16 [N][K].
// ---------------------------------------------------------------------------

typedef __attribute__((ext_vector_type(16))) __bf16 v16bf;
typedef __attribute__((ext_vector_type(8)))  __bf16 v8bf;
typedef __attribute__((ext_vector_type(8)))  float  v8f;

#define MTOK   32768      // B*T
#define HID    1024
#define DIN    1088       // 512 + 512 + 64
#define NTOK   16
#define TLEN   4096
#define NBATCH 8
#define LDT    40         // padded LDS row stride (bf16 elems): 80B, 16B aligned, bank-spread

__device__ __forceinline__ float gelu_exact(float x) {
    return 0.5f * x * (1.0f + erff(x * 0.70710678118654752f));
}

// issue one async 16B global->LDS copy (per-lane addresses, GV mode)
__device__ __forceinline__ void async_b128(const __bf16* gsrc, const __bf16* lds_dst) {
    uint32_t dst = (uint32_t)(uintptr_t)lds_dst;          // low 32 bits = LDS byte offset
    uint64_t src = (uint64_t)(uintptr_t)gsrc;
    asm volatile("global_load_async_to_lds_b128 %0, %1, off"
                 :: "v"(dst), "v"(src) : "memory");
}
__device__ __forceinline__ void wait_async0() {
    asm volatile("s_wait_asynccnt 0x0" ::: "memory");
}

// ---------------------------------------------------------------------------
// Tiled WMMA GEMM: out[M,N] = epilogue(A[M,K] @ W[K,N] + bias)
//   A = Abf (bf16, row-major, stride K), B = Wt (bf16, pre-transposed [N][K]);
//   both staged via async global->LDS b128 copies, double-buffered.
//   MODE: 0 = bias, 1 = bias+GELU, 2 = bias + residual R
//   WBF : 0 = fp32 out only, 1 = fp32 + bf16 out, 2 = bf16 out only
// Block: 256 threads (8 waves). Tile 128x128, BK=32. Wave: 64x32 (4x2 accs).
// ---------------------------------------------------------------------------
template <int MODE, int WBF>
__global__ __launch_bounds__(256) void gemm_wmma(
    const __bf16* __restrict__ Abf,
    const __bf16* __restrict__ Wt,
    const float* __restrict__ bias, const float* __restrict__ R,
    float* __restrict__ out, __bf16* __restrict__ outbf,
    int N, int K)
{
    __shared__ __align__(16) __bf16 lA[2][128 * LDT];   // lA[m][k]
    __shared__ __align__(16) __bf16 lB[2][128 * LDT];   // lB[n][k] (W transposed)

    const int tid  = threadIdx.x;
    const int n0   = blockIdx.x * 128;
    const int m0   = blockIdx.y * 128;
    const int wave = tid >> 5, lane = tid & 31;
    const int wrow = wave >> 2, wcol = wave & 3;        // 2x4 wave grid
    const int hl   = lane >> 4, l16 = lane & 15;

    v8f acc[4][2];
#pragma unroll
    for (int i = 0; i < 4; ++i)
#pragma unroll
        for (int j = 0; j < 2; ++j)
#pragma unroll
            for (int r = 0; r < 8; ++r) acc[i][j][r] = 0.0f;

    // per-thread async chunk assignment: 512 x 16B chunks per 8KB tile
    const int c0r = tid >> 2,          c0s = tid & 3;           // chunk tid
    const int c1r = (tid + 256) >> 2,  c1s = tid & 3;           // chunk tid+256

    auto stage = [&](int buf, int k0) {
        // B tile: lB[n][k] <- Wt[n0+n][k0 + seg*8 ..]
        async_b128(Wt + (size_t)(n0 + c0r) * K + k0 + c0s * 8, &lB[buf][c0r * LDT + c0s * 8]);
        async_b128(Wt + (size_t)(n0 + c1r) * K + k0 + c1s * 8, &lB[buf][c1r * LDT + c1s * 8]);
        // A tile: lA[r][k] <- Abf[m0+r][k0 + seg*8 ..]
        async_b128(Abf + (size_t)(m0 + c0r) * K + k0 + c0s * 8, &lA[buf][c0r * LDT + c0s * 8]);
        async_b128(Abf + (size_t)(m0 + c1r) * K + k0 + c1s * 8, &lA[buf][c1r * LDT + c1s * 8]);
    };

    stage(0, 0);
    const int nIter = K >> 5;
    for (int it = 0; it < nIter; ++it) {
        const int cur = it & 1;
        const int k0  = it << 5;
        wait_async0();          // my async copies into buf `cur` landed
        __syncthreads();        // everyone's copies landed
        if (k0 + 32 < K) stage(cur ^ 1, k0 + 32);   // DMA next slab during compute

        // ---- fragments per ISA VGPR layouts ----
        const __bf16* A_ = lA[cur];
        const __bf16* B_ = lB[cur];
        v16bf af[4], bfr[2];
#pragma unroll
        for (int i = 0; i < 4; ++i) {
            int row = wrow * 64 + i * 16 + l16;                         // M = l16
            const v8bf a0 = *(const v8bf*)&A_[row * LDT + hl * 8];      // K 0-7 / 8-15
            const v8bf a1 = *(const v8bf*)&A_[row * LDT + 16 + hl * 8]; // K 16-23 / 24-31
            af[i] = __builtin_shufflevector(a0, a1, 0,1,2,3,4,5,6,7,8,9,10,11,12,13,14,15);
        }
#pragma unroll
        for (int j = 0; j < 2; ++j) {
            int col = wcol * 32 + j * 16 + l16;                         // N = l16
            const v8bf b0 = *(const v8bf*)&B_[col * LDT + hl * 8];      // K 0-7 / 8-15
            const v8bf b1 = *(const v8bf*)&B_[col * LDT + 16 + hl * 8]; // K 16-23 / 24-31
            bfr[j] = __builtin_shufflevector(b0, b1, 0,1,2,3,4,5,6,7,8,9,10,11,12,13,14,15);
        }
#pragma unroll
        for (int i = 0; i < 4; ++i)
#pragma unroll
            for (int j = 0; j < 2; ++j)
                acc[i][j] = __builtin_amdgcn_wmma_f32_16x16x32_bf16(
                    false, af[i], false, bfr[j], (short)0, acc[i][j], false, false);
        __syncthreads();        // all waves done reading buf `cur`
    }

    // ---- epilogue ---------------------------------------------------------
#pragma unroll
    for (int i = 0; i < 4; ++i)
#pragma unroll
        for (int j = 0; j < 2; ++j) {
            const int gn = n0 + wcol * 32 + j * 16 + l16;
            const float bv = bias[gn];
#pragma unroll
            for (int r = 0; r < 8; ++r) {
                const int gm = m0 + wrow * 64 + i * 16 + hl * 8 + r;
                float v = acc[i][j][r] + bv;
                if constexpr (MODE == 1) v = gelu_exact(v);
                if constexpr (MODE == 2) v += R[(size_t)gm * N + gn];
                if constexpr (WBF != 2) out[(size_t)gm * N + gn] = v;
                if constexpr (WBF >= 1) outbf[(size_t)gm * N + gn] = (__bf16)v;
            }
        }
}

// pack concat(trace|semantic|prior) fp32 -> contiguous bf16 [MTOK, DIN]
__global__ __launch_bounds__(256) void pack_fused_kernel(
    const float* __restrict__ tr, const float* __restrict__ se,
    const float* __restrict__ pr, __bf16* __restrict__ fused)
{
    size_t t = (size_t)blockIdx.x * 256 + threadIdx.x;
    int row = (int)(t / DIN), col = (int)(t % DIN);
    float v;
    if (col < 512)       v = tr[(size_t)row * 512 + col];
    else if (col < 1024) v = se[(size_t)row * 512 + (col - 512)];
    else                 v = pr[(size_t)row * 64  + (col - 1024)];
    fused[t] = (__bf16)v;
}

// one-time fp32 -> bf16 transposed weight conversion: Wt[n*K + k] = W[k*N + n]
__global__ __launch_bounds__(256) void transpose_bf16_kernel(
    const float* __restrict__ W, __bf16* __restrict__ Wt, int K, int N)
{
    size_t t = (size_t)blockIdx.x * 256 + threadIdx.x;   // t = k*N + n (coalesced read)
    int n = (int)(t % N), k = (int)(t / N);
    Wt[(size_t)n * K + k] = (__bf16)W[t];
}

// ---------------------------------------------------------------------------
// Block-wide reductions (256 threads)
// ---------------------------------------------------------------------------
__device__ __forceinline__ float block_sum(float v, float* red) {
    int tid = threadIdx.x;
    red[tid] = v; __syncthreads();
    for (int s = 128; s > 0; s >>= 1) { if (tid < s) red[tid] += red[tid + s]; __syncthreads(); }
    float r = red[0]; __syncthreads();
    return r;
}
__device__ __forceinline__ float block_max(float v, float* red) {
    int tid = threadIdx.x;
    red[tid] = v; __syncthreads();
    for (int s = 128; s > 0; s >>= 1) { if (tid < s) red[tid] = fmaxf(red[tid], red[tid + s]); __syncthreads(); }
    float r = red[0]; __syncthreads();
    return r;
}

// LayerNorm over rows of HID; optional broadcast residual addq[(row%nmod)*HID];
// optional bf16 shadow copy of the output.
__global__ __launch_bounds__(256) void layernorm_kernel(
    const float* __restrict__ x, const float* __restrict__ addq,
    const float* __restrict__ g, const float* __restrict__ bta,
    float* __restrict__ out, __bf16* __restrict__ outbf, int nmod)
{
    __shared__ float red[256];
    const size_t row = blockIdx.x;
    const float* xr = x + row * HID;
    const float* qr = addq ? addq + (size_t)(row % nmod) * HID : nullptr;
    float v[4];
    float s = 0.0f;
#pragma unroll
    for (int i = 0; i < 4; ++i) {
        int c = threadIdx.x + 256 * i;
        v[i] = xr[c] + (qr ? qr[c] : 0.0f);
        s += v[i];
    }
    const float mu = block_sum(s, red) * (1.0f / HID);
    s = 0.0f;
#pragma unroll
    for (int i = 0; i < 4; ++i) { float d = v[i] - mu; s += d * d; }
    const float var = block_sum(s, red) * (1.0f / HID);
    const float inv = rsqrtf(var + 1e-5f);
#pragma unroll
    for (int i = 0; i < 4; ++i) {
        int c = threadIdx.x + 256 * i;
        float o = (v[i] - mu) * inv * g[c] + bta[c];
        out[row * HID + c] = o;
        if (outbf) outbf[row * HID + c] = (__bf16)o;
    }
}

// objectness = sigmoid(o1[tok] . W_o2 + b_o2); one wave32 per token.
__global__ __launch_bounds__(256) void objectness_kernel(
    const float* __restrict__ o1, const float* __restrict__ w,
    const float* __restrict__ b2, float* __restrict__ out)
{
    const int wave = threadIdx.x >> 5, lane = threadIdx.x & 31;
    const int tok = blockIdx.x * 8 + wave;
    const float* r = o1 + (size_t)tok * 512;
    float s = 0.0f;
#pragma unroll
    for (int i = 0; i < 16; ++i) { int c = lane + 32 * i; s += r[c] * w[c]; }
    for (int off = 16; off; off >>= 1) s += __shfl_xor(s, off, 32);
    if (lane == 0) out[tok] = 1.0f / (1.0f + expf(-(s + b2[0])));
}

// logits[b,n,t] = (q[n].seq[b,t]) * H^-0.5 + 1.5*obj[b,t]; one wave per token.
__global__ __launch_bounds__(256) void attn_logits_kernel(
    const float* __restrict__ seq, const float* __restrict__ tq,
    const float* __restrict__ obj, float* __restrict__ logits)
{
    const int wave = threadIdx.x >> 5, lane = threadIdx.x & 31;
    const int tok = blockIdx.x * 8 + wave;
    const int b = tok >> 12, t = tok & (TLEN - 1);
    float sv[32];
    const float* row = seq + (size_t)tok * HID;
#pragma unroll
    for (int i = 0; i < 32; ++i) sv[i] = row[lane + 32 * i];
    const float ob = 1.5f * obj[tok];
    for (int n = 0; n < NTOK; ++n) {
        const float* q = tq + (size_t)n * HID;
        float s = 0.0f;
#pragma unroll
        for (int i = 0; i < 32; ++i) s += sv[i] * q[lane + 32 * i];
        for (int off = 16; off; off >>= 1) s += __shfl_xor(s, off, 32);
        if (lane == 0)
            logits[((size_t)b * NTOK + n) * TLEN + t] = s * 0.03125f + ob;  // 1024^-0.5
    }
}

// in-place row softmax over T=4096; one block per (b,n).
__global__ __launch_bounds__(256) void softmax_kernel(float* __restrict__ p)
{
    __shared__ float red[256];
    float* row = p + (size_t)blockIdx.x * TLEN;
    float v[16];
    float mx = -1e30f;
#pragma unroll
    for (int i = 0; i < 16; ++i) { v[i] = row[threadIdx.x + 256 * i]; mx = fmaxf(mx, v[i]); }
    mx = block_max(mx, red);
    float s = 0.0f;
#pragma unroll
    for (int i = 0; i < 16; ++i) { v[i] = expf(v[i] - mx); s += v[i]; }
    const float inv = 1.0f / block_sum(s, red);
#pragma unroll
    for (int i = 0; i < 16; ++i) row[threadIdx.x + 256 * i] = v[i] * inv;
}

// tv[b,n,h] = sum_t attn[b,n,t]*seq[b,t,h]; block = (b, 256-wide h chunk).
__global__ __launch_bounds__(256) void token_values_kernel(
    const float* __restrict__ attn, const float* __restrict__ seq,
    float* __restrict__ tv)
{
    __shared__ float lat[NTOK][257];
    const int b = blockIdx.x >> 2;          // HID/256 = 4 chunks
    const int hcol = (blockIdx.x & 3) * 256 + threadIdx.x;
    float acc[NTOK];
#pragma unroll
    for (int n = 0; n < NTOK; ++n) acc[n] = 0.0f;
    for (int t0 = 0; t0 < TLEN; t0 += 256) {
        __syncthreads();
#pragma unroll
        for (int n = 0; n < NTOK; ++n)
            lat[n][threadIdx.x] = attn[((size_t)b * NTOK + n) * TLEN + t0 + threadIdx.x];
        __syncthreads();
        for (int tt = 0; tt < 256; ++tt) {
            const float s = seq[((size_t)b * TLEN + t0 + tt) * HID + hcol];
#pragma unroll
            for (int n = 0; n < NTOK; ++n) acc[n] += lat[n][tt] * s;
        }
    }
#pragma unroll
    for (int n = 0; n < NTOK; ++n)
        tv[((size_t)b * NTOK + n) * HID + hcol] = acc[n];
}

// ---------------------------------------------------------------------------
extern "C" void kernel_launch(void* const* d_in, const int* in_sizes, int n_in,
                              void* d_out, int out_size, void* d_ws, size_t ws_size,
                              hipStream_t stream) {
    const float* tr     = (const float*)d_in[0];
    const float* se     = (const float*)d_in[1];
    const float* pr     = (const float*)d_in[2];
    const float* W_in   = (const float*)d_in[3];
    const float* b_in   = (const float*)d_in[4];
    const float* ln_ig  = (const float*)d_in[5];
    const float* ln_ib  = (const float*)d_in[6];
    const float* W_m1   = (const float*)d_in[7];
    const float* b_m1   = (const float*)d_in[8];
    const float* W_m2   = (const float*)d_in[9];
    const float* b_m2   = (const float*)d_in[10];
    const float* W_o1   = (const float*)d_in[11];
    const float* b_o1   = (const float*)d_in[12];
    const float* W_o2   = (const float*)d_in[13];
    const float* b_o2   = (const float*)d_in[14];
    const float* tq     = (const float*)d_in[15];
    const float* ln_og  = (const float*)d_in[16];
    const float* ln_ob  = (const float*)d_in[17];

    float* out = (float*)d_out;
    float* out_state = out;                                       // 8*16*1024
    float* out_attn  = out + (size_t)NBATCH * NTOK * HID;         // 8*16*4096
    float* out_obj   = out_attn + (size_t)NBATCH * NTOK * TLEN;   // 8*4096

    const size_t fH = (size_t)MTOK * HID;          // 33.55M floats
    float*  ws     = (float*)d_ws;
    float*  h      = ws;                           // fp32 h  (later o1)
    float*  seq    = ws + fH;                      // fp32 seq
    float*  bfA    = ws + 2 * fH;                  // fH/2 floats: h_bf then seq_bf
    float*  bfB    = ws + 2 * fH + fH / 2;         // fH/2 floats: g_bf then tv
    float*  fusedA = ws + 3 * fH;                  // MTOK*DIN bf16 = 17.83M floats
    float*  wtarea = fusedA + ((size_t)MTOK * DIN) / 2;  // bf16 weights (~1.87M floats)

    __bf16* fused  = (__bf16*)fusedA;
    __bf16* h_bf   = (__bf16*)bfA;
    __bf16* seq_bf = (__bf16*)bfA;                 // reuse: h_bf dead after GEMM 2
    __bf16* g_bf   = (__bf16*)bfB;
    float*  tv     = bfB;                          // reuse: g_bf dead after GEMM 3
    float*  o1     = h;                            // reuse: h dead after GEMM 3

    __bf16* Wt_in = (__bf16*)wtarea;               // [HID][DIN]
    __bf16* Wt_m1 = Wt_in + (size_t)DIN * HID;     // [HID][HID]
    __bf16* Wt_m2 = Wt_m1 + (size_t)HID * HID;
    __bf16* Wt_o1 = Wt_m2 + (size_t)HID * HID;     // [512][HID]

    const dim3 blk(256);
    const dim3 gemm_grid(HID / 128, MTOK / 128);   // 8 x 256

    // 0) pre-pack: fused input -> bf16 [M, DIN]; weights -> bf16 transposed [N][K]
    pack_fused_kernel<<<((size_t)MTOK * DIN) / 256, blk, 0, stream>>>(tr, se, pr, fused);
    transpose_bf16_kernel<<<(DIN * HID) / 256, blk, 0, stream>>>(W_in, Wt_in, DIN, HID);
    transpose_bf16_kernel<<<(HID * HID) / 256, blk, 0, stream>>>(W_m1, Wt_m1, HID, HID);
    transpose_bf16_kernel<<<(HID * HID) / 256, blk, 0, stream>>>(W_m2, Wt_m2, HID, HID);
    transpose_bf16_kernel<<<(HID * 512) / 256, blk, 0, stream>>>(W_o1, Wt_o1, HID, 512);

    // 1) h = fused @ W_in + b_in  (K=1088; fp32 + bf16 out)
    gemm_wmma<0, 1><<<gemm_grid, blk, 0, stream>>>(
        fused, Wt_in, b_in, nullptr, h, h_bf, HID, DIN);
    // 2) g = GELU(h @ W_m1 + b_m1)   (bf16 out only; feeds GEMM 3)
    gemm_wmma<1, 2><<<gemm_grid, blk, 0, stream>>>(
        h_bf, Wt_m1, b_m1, nullptr, nullptr, g_bf, HID, HID);
    // 3) seq_pre = h + (g @ W_m2 + b_m2)
    gemm_wmma<2, 0><<<gemm_grid, blk, 0, stream>>>(
        g_bf, Wt_m2, b_m2, h, seq, nullptr, HID, HID);
    // 4) seq = LayerNorm(seq_pre) in place (+ bf16 shadow for GEMM 5)
    layernorm_kernel<<<MTOK, blk, 0, stream>>>(seq, nullptr, ln_ig, ln_ib, seq, seq_bf, NTOK);
    // 5) o1 = GELU(seq @ W_o1 + b_o1)   (N=512)
    gemm_wmma<1, 0><<<dim3(512 / 128, MTOK / 128), blk, 0, stream>>>(
        seq_bf, Wt_o1, b_o1, nullptr, o1, nullptr, 512, HID);
    // 6) objectness = sigmoid(o1 . W_o2 + b_o2)
    objectness_kernel<<<MTOK / 8, blk, 0, stream>>>(o1, W_o2, b_o2, out_obj);
    // 7) attention logits (scaled dot + objectness bias)
    attn_logits_kernel<<<MTOK / 8, blk, 0, stream>>>(seq, tq, out_obj, out_attn);
    // 8) softmax over T, in place in attn output region
    softmax_kernel<<<NBATCH * NTOK, blk, 0, stream>>>(out_attn);
    // 9) token values = attn @ seq
    token_values_kernel<<<NBATCH * (HID / 256), blk, 0, stream>>>(out_attn, seq, tv);
    // 10) state_tokens = LayerNorm(tq + tv)
    layernorm_kernel<<<NBATCH * NTOK, blk, 0, stream>>>(tv, tq, ln_og, ln_ob, out_state, nullptr, NTOK);
}
